// Local_21534966022847
// MI455X (gfx1250) — matrix operand order repaired and checked by reading, not compiled, not run on hardware
//
#include <hip/hip_runtime.h>
#include <hip/hip_bf16.h>
#include <cstddef>
#include <cstdint>

// ---------------------------------------------------------------------------
// Problem constants (match reference)
// ---------------------------------------------------------------------------
constexpr int B_    = 16;
constexpr int N_    = 4096;
constexpr int CIN_  = 64;
constexpr int S_    = 1024;
constexpr int KNN_  = 24;
constexpr int COUT_ = 128;
constexpr float EPS_ = 1e-5f;

// LDS activation layout: column-major [32 cols][ASZ halves], stride chosen so
// 16-lane x 32B vector LDS reads tile the banks (144 halves = 72 dwords, 8-bank
// shift per column).
constexpr int ASZ = 144;

typedef __attribute__((ext_vector_type(16))) _Float16 v16h;
typedef __attribute__((ext_vector_type(8)))  _Float16 v8h;
typedef __attribute__((ext_vector_type(8)))  float    v8f;

// ---------------------------------------------------------------------------
// WMMA helpers (CDNA5 wave32, V_WMMA_F32_16X16X32_F16)
// A fragment (16x32 f16):  lane m = lane&15;
//   half h: K = (lane>>4)*16 + h
// B fragment (32x16 f16): col n = lane&15; halves h -> K = (lane>>4)*16 + h
// C/D (16x16 f32): element (m,n): m = r + (lane>>4)*8, n = lane&15
// ---------------------------------------------------------------------------
__device__ __forceinline__ v16h load_w_frag(const float* __restrict__ W, int cdim,
                                            int row0, int kbase, int lane) {
  const int m     = lane & 15;
  const int half16 = (lane >> 4) << 4;
  v16h a;
#pragma unroll
  for (int h = 0; h < 16; ++h) {
    const int kl = half16 + h;
    a[h] = (_Float16)W[(size_t)(row0 + m) * cdim + kbase + kl];
  }
  return a;
}

template <int KSTEPS>
__device__ __forceinline__ void gemm_tiles(const v16h* wf, const _Float16* bin,
                                           int lane, v8f& acc0, v8f& acc1) {
  const int nl = lane & 15;
  const int kh = (lane >> 4) << 4;  // 0 or 16
  v8f z = {0.f, 0.f, 0.f, 0.f, 0.f, 0.f, 0.f, 0.f};
  acc0 = z;
  acc1 = z;
#pragma unroll
  for (int ks = 0; ks < KSTEPS; ++ks) {
    const int kb = (ks << 5) + kh;
    v16h b0 = *reinterpret_cast<const v16h*>(&bin[nl * ASZ + kb]);
    v16h b1 = *reinterpret_cast<const v16h*>(&bin[(nl + 16) * ASZ + kb]);
    acc0 = __builtin_amdgcn_wmma_f32_16x16x32_f16(false, wf[ks], false, b0,
                                                  (short)0, acc0, false, false);
    acc1 = __builtin_amdgcn_wmma_f32_16x16x32_f16(false, wf[ks], false, b1,
                                                  (short)0, acc1, false, false);
  }
}

// BN (+optional residual) + ReLU epilogue; stores f16 tile back to LDS
__device__ __forceinline__ void epilogue(v8f a0, v8f a1, int mtile, int lane,
                                         const float* sc, const float* sh,
                                         const _Float16* resid, _Float16* bout) {
  const int nl = lane & 15;
  const int rb = (mtile << 4) + ((lane >> 4) << 3);
  v8h r0, r1;
  if (resid) {
    r0 = *reinterpret_cast<const v8h*>(&resid[nl * ASZ + rb]);
    r1 = *reinterpret_cast<const v8h*>(&resid[(nl + 16) * ASZ + rb]);
  }
  v8h o0, o1;
#pragma unroll
  for (int r = 0; r < 8; ++r) {
    const int o = rb + r;
    const float s = sc[o], t = sh[o];
    float v0 = a0[r] * s + t;
    float v1 = a1[r] * s + t;
    if (resid) {
      v0 += (float)r0[r];
      v1 += (float)r1[r];
    }
    v0 = fmaxf(v0, 0.f);
    v1 = fmaxf(v1, 0.f);
    o0[r] = (_Float16)v0;
    o1[r] = (_Float16)v1;
  }
  *reinterpret_cast<v8h*>(&bout[nl * ASZ + rb]) = o0;
  *reinterpret_cast<v8h*>(&bout[(nl + 16) * ASZ + rb]) = o1;
}

// ---------------------------------------------------------------------------
// Kernel 1: farthest point sampling + new_xyz  (1 block / batch, 256 thr)
// ---------------------------------------------------------------------------
__global__ __launch_bounds__(256) void fps_kernel(const float* __restrict__ xyz,
                                                  int* __restrict__ fps,
                                                  float* __restrict__ outXYZ) {
  const int b = blockIdx.x;
  const int t = threadIdx.x;
  __shared__ float rv[256];
  __shared__ int   ri[256];
  __shared__ float cur[3];

  float px[16], py[16], pz[16], dist[16];
#pragma unroll
  for (int i = 0; i < 16; ++i) {
    const int n = t + (i << 8);
    const size_t o = ((size_t)b * N_ + n) * 3;
    px[i] = xyz[o + 0];
    py[i] = xyz[o + 1];
    pz[i] = xyz[o + 2];
    dist[i] = 1e10f;
  }

  int far = 0;
  for (int s = 0; s < S_; ++s) {
    if (t == (far & 255)) {
      const int i = far >> 8;
      cur[0] = px[i];
      cur[1] = py[i];
      cur[2] = pz[i];
    }
    __syncthreads();
    const float cx = cur[0], cy = cur[1], cz = cur[2];
    if (t == 0) {
      fps[b * S_ + s] = far;
      const size_t o = ((size_t)b * S_ + s) * 3;
      outXYZ[o + 0] = cx;
      outXYZ[o + 1] = cy;
      outXYZ[o + 2] = cz;
    }
    float bv = -1.f;
    int bi = 0;
#pragma unroll
    for (int i = 0; i < 16; ++i) {
      const float dx = px[i] - cx, dy = py[i] - cy, dz = pz[i] - cz;
      const float d = dx * dx + dy * dy + dz * dz;
      const float dm = fminf(dist[i], d);
      dist[i] = dm;
      if (dm > bv) {
        bv = dm;
        bi = t + (i << 8);
      }
    }
    rv[t] = bv;
    ri[t] = bi;
    __syncthreads();
    for (int off = 128; off > 0; off >>= 1) {
      if (t < off) {
        if (rv[t + off] > rv[t] ||
            (rv[t + off] == rv[t] && ri[t + off] < ri[t])) {
          rv[t] = rv[t + off];
          ri[t] = ri[t + off];
        }
      }
      __syncthreads();
    }
    far = ri[0];
    __syncthreads();
  }
}

// ---------------------------------------------------------------------------
// Kernel 2: KNN (24 smallest d2, ascending, stable) — 1 block / (b,s)
// ---------------------------------------------------------------------------
__global__ __launch_bounds__(256) void knn_kernel(const float* __restrict__ xyz,
                                                  const int* __restrict__ fps,
                                                  int* __restrict__ knn) {
  const int g = blockIdx.x;
  const int b = g >> 10;
  const int t = threadIdx.x;
  __shared__ float d2[N_];
  __shared__ float rv[256];
  __shared__ int   ri[256];

  const int sIdx = fps[g];
  const size_t qo = ((size_t)b * N_ + sIdx) * 3;
  const float qx = xyz[qo + 0], qy = xyz[qo + 1], qz = xyz[qo + 2];
  const float qq = qx * qx + qy * qy + qz * qz;

#pragma unroll
  for (int i = 0; i < 16; ++i) {
    const int n = t + (i << 8);
    const size_t o = ((size_t)b * N_ + n) * 3;
    const float x0 = xyz[o + 0], y0 = xyz[o + 1], z0 = xyz[o + 2];
    const float pp = x0 * x0 + y0 * y0 + z0 * z0;
    const float dot = qx * x0 + qy * y0 + qz * z0;
    d2[n] = qq + pp - 2.f * dot;
  }
  __syncthreads();

  for (int j = 0; j < KNN_; ++j) {
    float bv = 3.4e38f;
    int bi = N_;
#pragma unroll
    for (int i = 0; i < 16; ++i) {
      const int n = t + (i << 8);
      const float v = d2[n];
      if (v < bv) {
        bv = v;
        bi = n;
      }
    }
    rv[t] = bv;
    ri[t] = bi;
    __syncthreads();
    for (int off = 128; off > 0; off >>= 1) {
      if (t < off) {
        if (rv[t + off] < rv[t] ||
            (rv[t + off] == rv[t] && ri[t + off] < ri[t])) {
          rv[t] = rv[t + off];
          ri[t] = ri[t + off];
        }
      }
      __syncthreads();
    }
    const int w = ri[0];
    if (t == 0) {
      knn[(size_t)g * KNN_ + j] = w;
      d2[w] = 3.0e38f;
    }
    __syncthreads();
  }
}

// ---------------------------------------------------------------------------
// Kernel 3: PreExtraction (gather + conv/BN/ReLU + 2 res blocks + maxpool)
// 256 threads = 8 waves; wave w owns M-tile w; WMMA f32<-f16 16x16x32.
// ---------------------------------------------------------------------------
__global__ __launch_bounds__(256) void pre_kernel(
    const float* __restrict__ x, const int* __restrict__ fps,
    const int* __restrict__ knn, const float* __restrict__ tw,
    const float* __restrict__ tbn, const float* __restrict__ pw1,
    const float* __restrict__ pbn1, const float* __restrict__ pw2,
    const float* __restrict__ pbn2, float* __restrict__ hpre) {
  const int t = threadIdx.x;
  const int lane = t & 31;
  const int mtile = t >> 5;

  __shared__ __align__(32) _Float16 bufA[32 * ASZ];
  __shared__ __align__(32) _Float16 bufB[32 * ASZ];
  __shared__ __align__(32) _Float16 bufC[32 * ASZ];
  __shared__ float bnS[5][COUT_];
  __shared__ float bnB[5][COUT_];
  __shared__ int   sg[KNN_];
  __shared__ float anchor[CIN_];

  // ---- preload all weight A-fragments into registers (shared by all groups)
  v16h wT[2], w11[4], w12[4], w21[4], w22[4];
#pragma unroll
  for (int ks = 0; ks < 2; ++ks)
    wT[ks] = load_w_frag(tw, CIN_, mtile * 16, ks * 32, lane);
#pragma unroll
  for (int ks = 0; ks < 4; ++ks) {
    w11[ks] = load_w_frag(pw1, COUT_, mtile * 16, ks * 32, lane);
    w12[ks] = load_w_frag(pw2, COUT_, mtile * 16, ks * 32, lane);
    w21[ks] = load_w_frag(pw1 + COUT_ * COUT_, COUT_, mtile * 16, ks * 32, lane);
    w22[ks] = load_w_frag(pw2 + COUT_ * COUT_, COUT_, mtile * 16, ks * 32, lane);
  }

  // ---- fold BN params: scale = g*rsqrt(v+eps), shift = b - m*scale
  if (t < COUT_) {
    auto stage = [&](int l, const float* p) {
      const float g = p[t], bb = p[COUT_ + t];
      const float mm = p[2 * COUT_ + t], vv = p[3 * COUT_ + t];
      const float scl = g * rsqrtf(vv + EPS_);
      bnS[l][t] = scl;
      bnB[l][t] = bb - mm * scl;
    };
    stage(0, tbn);
    stage(1, pbn1);
    stage(2, pbn2);
    stage(3, pbn1 + 4 * COUT_);
    stage(4, pbn2 + 4 * COUT_);
  }

  for (int g = blockIdx.x; g < B_ * S_; g += gridDim.x) {
    const int b = g >> 10;
    const int scol = g & 1023;
    __syncthreads();  // buffers from previous group fully consumed
    if (t < KNN_) sg[t] = knn[(size_t)g * KNN_ + t];
    if (t >= 64 && t < 64 + CIN_) {
      const int c = t - 64;
      anchor[c] = x[((size_t)b * CIN_ + c) * N_ + fps[g]];
    }
    const int gn = g + gridDim.x;
    if (gn < B_ * S_)
      __builtin_prefetch((const void*)(knn + (size_t)gn * KNN_), 0, 1);
    __syncthreads();

    // gather: bufA[n][c] = x[b][c][idx_n] - anchor[c]  (cols 24..31 zero)
    for (int e = t; e < CIN_ * 32; e += 256) {
      const int c = e >> 5;
      const int n = e & 31;
      float v = 0.f;
      if (n < KNN_) v = x[((size_t)b * CIN_ + c) * N_ + sg[n]] - anchor[c];
      bufA[n * ASZ + c] = (_Float16)v;
    }
    __syncthreads();

    v8f a0, a1;
    // L1: transfer conv (128x64) + BN + ReLU
    gemm_tiles<2>(wT, bufA, lane, a0, a1);
    epilogue(a0, a1, mtile, lane, bnS[0], bnB[0], nullptr, bufB);
    __syncthreads();
    // L2: block1 conv1 + BN + ReLU
    gemm_tiles<4>(w11, bufB, lane, a0, a1);
    epilogue(a0, a1, mtile, lane, bnS[1], bnB[1], nullptr, bufC);
    __syncthreads();
    // L3: block1 conv2 + BN, + h, ReLU
    gemm_tiles<4>(w12, bufC, lane, a0, a1);
    epilogue(a0, a1, mtile, lane, bnS[2], bnB[2], bufB, bufA);
    __syncthreads();
    // L4: block2 conv1 + BN + ReLU
    gemm_tiles<4>(w21, bufA, lane, a0, a1);
    epilogue(a0, a1, mtile, lane, bnS[3], bnB[3], nullptr, bufC);
    __syncthreads();
    // L5: block2 conv2 + BN, + h, ReLU, maxpool over n<24
    gemm_tiles<4>(w22, bufC, lane, a0, a1);
    {
      const int nl = lane & 15;
      const int rb = (mtile << 4) + ((lane >> 4) << 3);
      v8h r0 = *reinterpret_cast<const v8h*>(&bufA[nl * ASZ + rb]);
      v8h r1 = *reinterpret_cast<const v8h*>(&bufA[(nl + 16) * ASZ + rb]);
#pragma unroll
      for (int r = 0; r < 8; ++r) {
        const int o = rb + r;
        const float sc = bnS[4][o], sh = bnB[4][o];
        const float v0 = fmaxf(a0[r] * sc + sh + (float)r0[r], 0.f);
        const float v1 = fmaxf(a1[r] * sc + sh + (float)r1[r], 0.f);
        float m = (nl < 8) ? fmaxf(v0, v1) : v0;  // cols 16..23 only
        m = fmaxf(m, __shfl_xor(m, 1, 32));
        m = fmaxf(m, __shfl_xor(m, 2, 32));
        m = fmaxf(m, __shfl_xor(m, 4, 32));
        m = fmaxf(m, __shfl_xor(m, 8, 32));
        if (nl == 0) hpre[((size_t)b * COUT_ + o) * S_ + scol] = m;
      }
    }
  }
}

// ---------------------------------------------------------------------------
// Kernel 4: PosExtraction (2 res blocks on [B, COUT, S]) — block per
// (batch, 32-column tile). Input tile staged with CDNA5 async LDS DMA
// (GLOBAL_LOAD_ASYNC_TO_LDS_B128, ASYNCcnt), then f16-converted for WMMA.
// ---------------------------------------------------------------------------
__global__ __launch_bounds__(256) void pos_kernel(
    const float* __restrict__ hpre, const float* __restrict__ pw1,
    const float* __restrict__ pbn1, const float* __restrict__ pw2,
    const float* __restrict__ pbn2, float* __restrict__ outH) {
  const int t = threadIdx.x;
  const int lane = t & 31;
  const int mtile = t >> 5;
  const int b = blockIdx.x >> 5;
  const int c0 = (blockIdx.x & 31) << 5;

  __shared__ __align__(32) _Float16 bufA[32 * ASZ];
  __shared__ __align__(32) _Float16 bufB[32 * ASZ];
  __shared__ __align__(32) _Float16 bufC[32 * ASZ];
  __shared__ __align__(16) float stage[COUT_ * 32];  // fp32 input tile (16KB)
  __shared__ float bnS[4][COUT_];
  __shared__ float bnB[4][COUT_];

  // ---- async DMA of the [128 x 32] fp32 input tile into LDS (ASYNCcnt path)
  {
    const unsigned ldsbase = (unsigned)(uintptr_t)(&stage[0]);
    for (int q = t; q < COUT_ * 8; q += 256) {  // 16-byte chunks
      const int c = q >> 3;
      const int f = (q & 7) << 2;
      const float* gptr = hpre + ((size_t)b * COUT_ + c) * S_ + c0 + f;
      const unsigned dst = ldsbase + (unsigned)((c * 32 + f) * 4);
      asm volatile("global_load_async_to_lds_b128 %0, %1, off"
                   :
                   : "v"(dst), "v"(gptr)
                   : "memory");
    }
    asm volatile("s_wait_asynccnt 0" ::: "memory");
  }
  __syncthreads();

  v16h wa[4], wb[4], wc[4], wd[4];
#pragma unroll
  for (int ks = 0; ks < 4; ++ks) {
    wa[ks] = load_w_frag(pw1, COUT_, mtile * 16, ks * 32, lane);
    wb[ks] = load_w_frag(pw2, COUT_, mtile * 16, ks * 32, lane);
    wc[ks] = load_w_frag(pw1 + COUT_ * COUT_, COUT_, mtile * 16, ks * 32, lane);
    wd[ks] = load_w_frag(pw2 + COUT_ * COUT_, COUT_, mtile * 16, ks * 32, lane);
  }
  if (t < COUT_) {
    auto stg = [&](int l, const float* p) {
      const float g = p[t], bb = p[COUT_ + t];
      const float mm = p[2 * COUT_ + t], vv = p[3 * COUT_ + t];
      const float scl = g * rsqrtf(vv + EPS_);
      bnS[l][t] = scl;
      bnB[l][t] = bb - mm * scl;
    };
    stg(0, pbn1);
    stg(1, pbn2);
    stg(2, pbn1 + 4 * COUT_);
    stg(3, pbn2 + 4 * COUT_);
  }

  // convert staged tile: bufA[n][c] = (f16) stage[c][n]
  for (int e = t; e < COUT_ * 32; e += 256) {
    const int c = e >> 5;
    const int n = e & 31;
    bufA[n * ASZ + c] = (_Float16)stage[c * 32 + n];
  }
  __syncthreads();

  v8f a0, a1;
  // block1 conv1 + BN + ReLU
  gemm_tiles<4>(wa, bufA, lane, a0, a1);
  epilogue(a0, a1, mtile, lane, bnS[0], bnB[0], nullptr, bufB);
  __syncthreads();
  // block1 conv2 + BN, + h, ReLU  -> h1
  gemm_tiles<4>(wb, bufB, lane, a0, a1);
  epilogue(a0, a1, mtile, lane, bnS[1], bnB[1], bufA, bufC);
  __syncthreads();
  // block2 conv1 + BN + ReLU
  gemm_tiles<4>(wc, bufC, lane, a0, a1);
  epilogue(a0, a1, mtile, lane, bnS[2], bnB[2], nullptr, bufB);
  __syncthreads();
  // block2 conv2 + BN, + h1, ReLU -> final fp32 output
  gemm_tiles<4>(wd, bufB, lane, a0, a1);
  {
    const int nl = lane & 15;
    const int rb = (mtile << 4) + ((lane >> 4) << 3);
    v8h r0 = *reinterpret_cast<const v8h*>(&bufC[nl * ASZ + rb]);
    v8h r1 = *reinterpret_cast<const v8h*>(&bufC[(nl + 16) * ASZ + rb]);
#pragma unroll
    for (int r = 0; r < 8; ++r) {
      const int o = rb + r;
      const float sc = bnS[3][o], sh = bnB[3][o];
      const float v0 = fmaxf(a0[r] * sc + sh + (float)r0[r], 0.f);
      const float v1 = fmaxf(a1[r] * sc + sh + (float)r1[r], 0.f);
      outH[((size_t)b * COUT_ + o) * S_ + c0 + nl] = v0;
      outH[((size_t)b * COUT_ + o) * S_ + c0 + 16 + nl] = v1;
    }
  }
}

// ---------------------------------------------------------------------------
// Launcher
// ---------------------------------------------------------------------------
extern "C" void kernel_launch(void* const* d_in, const int* in_sizes, int n_in,
                              void* d_out, int out_size, void* d_ws,
                              size_t ws_size, hipStream_t stream) {
  (void)in_sizes;
  (void)n_in;
  (void)out_size;
  (void)ws_size;
  const float* xyz     = (const float*)d_in[0];
  const float* x       = (const float*)d_in[1];
  const float* tw      = (const float*)d_in[2];
  const float* tbn     = (const float*)d_in[3];
  const float* pre_w1  = (const float*)d_in[4];
  const float* pre_bn1 = (const float*)d_in[5];
  const float* pre_w2  = (const float*)d_in[6];
  const float* pre_bn2 = (const float*)d_in[7];
  const float* pos_w1  = (const float*)d_in[8];
  const float* pos_bn1 = (const float*)d_in[9];
  const float* pos_w2  = (const float*)d_in[10];
  const float* pos_bn2 = (const float*)d_in[11];

  float* out    = (float*)d_out;
  float* outXYZ = out;                          // [B,S,3]
  float* outH   = out + (size_t)B_ * S_ * 3;    // [B,COUT,S]

  int*   fps  = (int*)d_ws;                     // [B*S]
  int*   knn  = fps + B_ * S_;                  // [B*S*K]
  float* hpre = (float*)(knn + (size_t)B_ * S_ * KNN_);  // [B,COUT,S]

  fps_kernel<<<B_, 256, 0, stream>>>(xyz, fps, outXYZ);
  knn_kernel<<<B_ * S_, 256, 0, stream>>>(xyz, fps, knn);
  pre_kernel<<<1024, 256, 0, stream>>>(x, fps, knn, tw, tbn, pre_w1, pre_bn1,
                                       pre_w2, pre_bn2, hpre);
  pos_kernel<<<B_ * (S_ / 32), 256, 0, stream>>>(hpre, pos_w1, pos_bn1, pos_w2,
                                                 pos_bn2, outH);
}
